// GatingAttention_55310588838176
// MI455X (gfx1250) — compile-verified, matrix operand order checked by reading, not run on hardware
//
#include <hip/hip_runtime.h>

// ---------------------------------------------------------------------------
// Types for CDNA5 WMMA (wave32): v16bf A/B fragments, v8f C/D accumulators
// ---------------------------------------------------------------------------
typedef __bf16 bf16;
typedef __attribute__((ext_vector_type(16))) __bf16 v16bf;
typedef __attribute__((ext_vector_type(8)))  float  v8f;
typedef __attribute__((ext_vector_type(4)))  unsigned int u32x4;

union V16BF { v16bf v; bf16 e[16]; u32x4 q[2]; };
union V8F   { v8f v;   float e[8]; };

// A-frag (16x32, M x K): lanes 0-15 hold row M=lane, K = 0..7 (elems 0..7) and
// K=16..23 (elems 8..15); lanes 16-31 hold K=8..15 / 24..31.  B-frag mirrors
// this with lane = N column, K striped identically.  Both reduce to two
// contiguous 16-byte loads per lane from K-contiguous memory.
__device__ __forceinline__ v16bf load_frag(const bf16* rowptr, int hi) {
  V16BF f;
  f.q[0] = *(const u32x4*)(rowptr + 8 * hi);
  f.q[1] = *(const u32x4*)(rowptr + 16 + 8 * hi);
  return f.v;
}

__device__ __forceinline__ v8f wmma_bf16(v16bf a, v16bf b, v8f c) {
  // (neg_a, A, neg_b, B, c_mod, C, reuse_a, reuse_b)
  return __builtin_amdgcn_wmma_f32_16x16x32_bf16(false, a, false, b,
                                                 (short)0, c, false, false);
}

__device__ __forceinline__ bf16 f2bf(float x) { return (bf16)x; }
__device__ __forceinline__ float bf2f(bf16 x) { return (float)x; }

// CDNA5 async global->LDS copy (ASYNCcnt-tracked). Per-lane 16B transfer:
//   LDS[vdst_lane_addr] = MEM[vaddr_lane].b128
__device__ __forceinline__ void async_g2l_b128(void* lds_ptr, const void* g) {
  unsigned loff = (unsigned)(uintptr_t)lds_ptr;           // low 32 = LDS addr
  unsigned long long ga = (unsigned long long)(uintptr_t)g;
  asm volatile("global_load_async_to_lds_b128 %0, %1, off"
               :: "v"(loff), "v"(ga)
               : "memory");
}
__device__ __forceinline__ void wait_async0() {
  asm volatile("s_wait_asynccnt 0x0" ::: "memory");
}

// Problem constants
constexpr int   Bn    = 256;
constexpr int   Sn    = 256;
constexpr int   Hn    = 8;
constexpr int   NROW  = Bn * Sn;           // 65536 rows of the fused (b,s) axis
constexpr float QSCALE = 0.17677669529663687f; // 32^-0.5
constexpr int   PSTR  = 264;  // P LDS row stride (bf16 elems), breaks 256B bank period
constexpr int   KSTR  = 40;   // K LDS row stride (32 kd + 8 pad)
constexpr int   VSTR  = 264;  // V^T LDS row stride (256 s + 8 pad)
constexpr int   P_ELEMS = 256 * PSTR;            // 67584
constexpr int   K_OFF   = P_ELEMS;               // K tile after P
constexpr int   V_OFF   = K_OFF + 256 * KSTR;    // V tile after K
constexpr int   LDS_ELEMS = V_OFF + 32 * VSTR;   // total bf16 elements

// ---------------------------------------------------------------------------
// Kernel 0: convert + transpose the five 256x256 f32 weight matrices to bf16
// W^T layout [n][k] so WMMA B-fragments are contiguous 16B loads.
// ---------------------------------------------------------------------------
__global__ __launch_bounds__(256) void prep_weights(
    const float* __restrict__ Wq, const float* __restrict__ Wk,
    const float* __restrict__ Wv, const float* __restrict__ Wg,
    const float* __restrict__ Wo,
    bf16* __restrict__ WqT, bf16* __restrict__ WkT, bf16* __restrict__ WvT,
    bf16* __restrict__ WgT, bf16* __restrict__ WoT) {
  int gid = blockIdx.x * 256 + threadIdx.x;       // 0 .. 5*65536-1
  int w   = gid >> 16;
  int idx = gid & 65535;
  int k = idx >> 8, n = idx & 255;
  const float* S[5] = {Wq, Wk, Wv, Wg, Wo};
  bf16*        D[5] = {WqT, WkT, WvT, WgT, WoT};
  D[w][n * 256 + k] = f2bf(S[w][idx]);
}

// ---------------------------------------------------------------------------
// Kernel 1: fused Q/K/V/gate projections.
//   grid = (1024, 4): x = 64-row tile of the 65536x256 input, y = projection id
//   block = 256 (8 waves); wave tile = 16 rows x 128 cols (8 WMMA accum tiles)
//   p==0: q * 32^-0.5 -> qb[b,s,hc]      p==1: k -> kb[b,s,hc]
//   p==2: v -> vT[b,hc,s] (transposed)   p==3: sigmoid(g + gb) -> gateT[b,hc,s]
// ---------------------------------------------------------------------------
__global__ __launch_bounds__(256) void proj_kernel(
    const float* __restrict__ q_data, const float* __restrict__ m_data,
    const bf16* __restrict__ WqT, const bf16* __restrict__ WkT,
    const bf16* __restrict__ WvT, const bf16* __restrict__ WgT,
    const float* __restrict__ gating_b,
    bf16* __restrict__ qb, bf16* __restrict__ kb,
    bf16* __restrict__ vT, bf16* __restrict__ gateT) {
  const int p    = blockIdx.y;
  const int tid  = threadIdx.x;
  const int wave = tid >> 5, lane = tid & 31;
  const int l    = lane & 15, hi = lane >> 4;
  const int wr   = wave & 3,  wc = wave >> 2;
  const int m0   = blockIdx.x * 64 + wr * 16;   // row of fused (b,s)
  const int n0   = wc * 128;                    // output column base

  const float* X  = (p == 0 || p == 3) ? q_data : m_data;
  const bf16*  WT = (p == 0) ? WqT : (p == 1) ? WkT : (p == 2) ? WvT : WgT;

  V8F acc[8];
#pragma unroll
  for (int t = 0; t < 8; ++t)
#pragma unroll
    for (int j = 0; j < 8; ++j) acc[t].e[j] = 0.0f;

  const float* xrow = X + (size_t)(m0 + l) * 256;
#pragma unroll
  for (int kc = 0; kc < 8; ++kc) {
    // A fragment: convert this lane's f32 row chunk to bf16 on the fly
    V16BF a;
    const float* xr = xrow + 32 * kc + 8 * hi;
#pragma unroll
    for (int i = 0; i < 8; ++i) {
      a.e[i]     = f2bf(xr[i]);
      a.e[i + 8] = f2bf(xr[16 + i]);
    }
#pragma unroll
    for (int t = 0; t < 8; ++t) {
      const bf16* wrow = WT + (size_t)(n0 + 16 * t + l) * 256 + 32 * kc;
      acc[t].v = wmma_bf16(a.v, load_frag(wrow, hi), acc[t].v);
    }
  }

  // Fused epilogues. C layout: lane holds col = tile_n + l, rows m0+8*hi+j.
#pragma unroll
  for (int t = 0; t < 8; ++t) {
    const int col = n0 + 16 * t + l;
    if (p == 0) {
#pragma unroll
      for (int j = 0; j < 8; ++j) {
        int r = m0 + 8 * hi + j;
        qb[(size_t)r * 256 + col] = f2bf(acc[t].e[j] * QSCALE);
      }
    } else if (p == 1) {
#pragma unroll
      for (int j = 0; j < 8; ++j) {
        int r = m0 + 8 * hi + j;
        kb[(size_t)r * 256 + col] = f2bf(acc[t].e[j]);
      }
    } else if (p == 2) {
#pragma unroll
      for (int j = 0; j < 8; ++j) {
        int r = m0 + 8 * hi + j;
        int b = r >> 8, s = r & 255;
        vT[((size_t)(b * 256 + col)) * 256 + s] = f2bf(acc[t].e[j]);
      }
    } else {
      const float gb = gating_b[col];
#pragma unroll
      for (int j = 0; j < 8; ++j) {
        int r = m0 + 8 * hi + j;
        int b = r >> 8, s = r & 255;
        float v = acc[t].e[j] + gb;
        v = 1.0f / (1.0f + __expf(-v));
        gateT[((size_t)(b * 256 + col)) * 256 + s] = f2bf(v);
      }
    }
  }
}

// ---------------------------------------------------------------------------
// Kernel 2: attention core, one workgroup per (b,h).
//   Async-stage K (256x32) and V^T (32x256) into padded LDS once per WG
//   (global_load_async_to_lds_b128 + s_wait_asynccnt), then:
//   logits via Q x K^T WMMAs with bias folded into C; full-row softmax via
//   shfl_xor; normalized P (bf16) staged in LDS; P x V^T WMMAs; gate; store.
// ---------------------------------------------------------------------------
__global__ __launch_bounds__(256) void attn_kernel(
    const bf16* __restrict__ qb, const bf16* __restrict__ kb,
    const bf16* __restrict__ vT, const bf16* __restrict__ gateT,
    const float* __restrict__ bias, const float* __restrict__ nbias,
    bf16* __restrict__ wag) {
  extern __shared__ bf16 Pl[];     // [256][PSTR] + K tile + V tile
  bf16* Kl = Pl + K_OFF;           // [256][KSTR] keys x kd
  bf16* Vl = Pl + V_OFF;           // [32][VSTR]  vd x keys
  const int b = blockIdx.x >> 3, h = blockIdx.x & 7;
  const int tid  = threadIdx.x;
  const int wave = tid >> 5, lane = tid & 31;
  const int l    = lane & 15, hi = lane >> 4;
  const int q0w  = wave * 32;

  // ---- Async prologue: K[b,h] and V^T[b,h] -> padded LDS tiles ----
  {
    const bf16* kg = kb + ((size_t)b * 256) * 256 + h * 32;       // 64B rows
    const bf16* vg = vT + ((size_t)(b * 256 + h * 32)) * 256;     // 512B rows
#pragma unroll
    for (int it = 0; it < 4; ++it) {        // K: 256 rows x 4 chunks of 16B
      int idx = it * 256 + tid;
      int row = idx >> 2, ch = idx & 3;
      async_g2l_b128(Kl + row * KSTR + ch * 8, kg + (size_t)row * 256 + ch * 8);
    }
#pragma unroll
    for (int it = 0; it < 4; ++it) {        // V: 32 rows x 32 chunks of 16B
      int idx = it * 256 + tid;
      int row = idx >> 5, ch = idx & 31;
      async_g2l_b128(Vl + row * VSTR + ch * 8, vg + (size_t)row * 256 + ch * 8);
    }
    wait_async0();
  }
  __syncthreads();

  // ---------------- Phase 1: logits + softmax + P -> LDS ----------------
#pragma unroll
  for (int qt = 0; qt < 2; ++qt) {
    const int q0 = q0w + 16 * qt;
    const v16bf qf =
        load_frag(qb + ((size_t)(b * 256 + q0 + l)) * 256 + h * 32, hi);

    V8F acc[16];
#pragma unroll
    for (int kt = 0; kt < 16; ++kt) {
      const int key = kt * 16 + l;
      const v16bf kf = load_frag(Kl + key * KSTR, hi);
      V8F c;
      const float bk = bias[b * 256 + key];
      const float* nb =
          nbias + ((size_t)(h * 256 + q0 + 8 * hi)) * 256 + key;
#pragma unroll
      for (int j = 0; j < 8; ++j) c.e[j] = bk + nb[(size_t)j * 256];
      acc[kt].v = wmma_bf16(qf, kf, c.v);
    }

    // row max across 16 key-tiles then across the 16 lanes of this half
    float m[8];
#pragma unroll
    for (int j = 0; j < 8; ++j) m[j] = acc[0].e[j];
#pragma unroll
    for (int kt = 1; kt < 16; ++kt)
#pragma unroll
      for (int j = 0; j < 8; ++j) m[j] = fmaxf(m[j], acc[kt].e[j]);
#pragma unroll
    for (int off = 1; off <= 8; off <<= 1)
#pragma unroll
      for (int j = 0; j < 8; ++j)
        m[j] = fmaxf(m[j], __shfl_xor(m[j], off, 32));

    float ssum[8];
#pragma unroll
    for (int j = 0; j < 8; ++j) ssum[j] = 0.0f;
#pragma unroll
    for (int kt = 0; kt < 16; ++kt)
#pragma unroll
      for (int j = 0; j < 8; ++j) {
        float ex = __expf(acc[kt].e[j] - m[j]);
        acc[kt].e[j] = ex;
        ssum[j] += ex;
      }
#pragma unroll
    for (int off = 1; off <= 8; off <<= 1)
#pragma unroll
      for (int j = 0; j < 8; ++j) ssum[j] += __shfl_xor(ssum[j], off, 32);

    float inv[8];
#pragma unroll
    for (int j = 0; j < 8; ++j) inv[j] = 1.0f / ssum[j];

#pragma unroll
    for (int kt = 0; kt < 16; ++kt) {
      const int key = kt * 16 + l;
#pragma unroll
      for (int j = 0; j < 8; ++j)
        Pl[(q0 + 8 * hi + j) * PSTR + key] = f2bf(acc[kt].e[j] * inv[j]);
    }
  }
  __syncthreads();

  // ---------------- Phase 2: WA = P x V, gate, store ----------------
  V8F wacc[2][2];
#pragma unroll
  for (int qt = 0; qt < 2; ++qt)
#pragma unroll
    for (int nt = 0; nt < 2; ++nt)
#pragma unroll
      for (int j = 0; j < 8; ++j) wacc[qt][nt].e[j] = 0.0f;

#pragma unroll
  for (int kc = 0; kc < 8; ++kc) {
    v16bf pa[2], vb[2];
#pragma unroll
    for (int qt = 0; qt < 2; ++qt)
      pa[qt] = load_frag(Pl + (q0w + 16 * qt + l) * PSTR + 32 * kc, hi);
#pragma unroll
    for (int nt = 0; nt < 2; ++nt)
      vb[nt] = load_frag(Vl + (nt * 16 + l) * VSTR + 32 * kc, hi);
#pragma unroll
    for (int qt = 0; qt < 2; ++qt)
#pragma unroll
      for (int nt = 0; nt < 2; ++nt)
        wacc[qt][nt].v = wmma_bf16(pa[qt], vb[nt], wacc[qt][nt].v);
  }

#pragma unroll
  for (int qt = 0; qt < 2; ++qt) {
    const int q0 = q0w + 16 * qt;
#pragma unroll
    for (int nt = 0; nt < 2; ++nt) {
      const int vd = nt * 16 + l;
      const int hc = h * 32 + vd;
      const bf16* gp =
          gateT + ((size_t)(b * 256 + hc)) * 256 + q0 + 8 * hi;
#pragma unroll
      for (int j = 0; j < 8; ++j) {
        float g = bf2f(gp[j]);
        wag[((size_t)(b * 256 + q0 + 8 * hi + j)) * 256 + hc] =
            f2bf(wacc[qt][nt].e[j] * g);
      }
    }
  }
}

// ---------------------------------------------------------------------------
// Kernel 3: output projection  out[65536,256] = wag x Wo + output_b  (f32 out)
// ---------------------------------------------------------------------------
__global__ __launch_bounds__(256) void outproj_kernel(
    const bf16* __restrict__ wag, const bf16* __restrict__ WoT,
    const float* __restrict__ out_b, float* __restrict__ out) {
  const int tid  = threadIdx.x;
  const int wave = tid >> 5, lane = tid & 31;
  const int l    = lane & 15, hi = lane >> 4;
  const int wr   = wave & 3,  wc = wave >> 2;
  const int m0   = blockIdx.x * 64 + wr * 16;
  const int n0   = wc * 128;

  V8F acc[8];
#pragma unroll
  for (int t = 0; t < 8; ++t)
#pragma unroll
    for (int j = 0; j < 8; ++j) acc[t].e[j] = 0.0f;

#pragma unroll
  for (int kc = 0; kc < 8; ++kc) {
    const v16bf a = load_frag(wag + (size_t)(m0 + l) * 256 + 32 * kc, hi);
#pragma unroll
    for (int t = 0; t < 8; ++t) {
      const bf16* wrow = WoT + (size_t)(n0 + 16 * t + l) * 256 + 32 * kc;
      acc[t].v = wmma_bf16(a, load_frag(wrow, hi), acc[t].v);
    }
  }

#pragma unroll
  for (int t = 0; t < 8; ++t) {
    const int col = n0 + 16 * t + l;
    const float ob = out_b[col];
#pragma unroll
    for (int j = 0; j < 8; ++j)
      out[(size_t)(m0 + 8 * hi + j) * 256 + col] = acc[t].e[j] + ob;
  }
}

// ---------------------------------------------------------------------------
// Host-side launch
// ---------------------------------------------------------------------------
extern "C" void kernel_launch(void* const* d_in, const int* in_sizes, int n_in,
                              void* d_out, int out_size, void* d_ws,
                              size_t ws_size, hipStream_t stream) {
  const float* q_data = (const float*)d_in[0];
  const float* m_data = (const float*)d_in[1];
  const float* bias   = (const float*)d_in[2];
  const float* nbias  = (const float*)d_in[3];
  const float* Wq     = (const float*)d_in[4];
  const float* Wk     = (const float*)d_in[5];
  const float* Wv     = (const float*)d_in[6];
  const float* Wg     = (const float*)d_in[7];
  const float* gbias  = (const float*)d_in[8];
  const float* Wo     = (const float*)d_in[9];
  const float* ob     = (const float*)d_in[10];
  float* out = (float*)d_out;

  char* ws = (char*)d_ws;
  const size_t SZ = (size_t)NROW * 256 * sizeof(bf16);  // 32 MiB per buffer
  bf16* qb    = (bf16*)(ws + 0 * SZ);   // [b,s,hc]  scaled Q
  bf16* kb    = (bf16*)(ws + 1 * SZ);   // [b,s,hc]  K
  bf16* vT    = (bf16*)(ws + 2 * SZ);   // [b,hc,s]  V transposed per head
  bf16* gateT = (bf16*)(ws + 3 * SZ);   // [b,hc,s]  sigmoid gate transposed
  bf16* wag   = (bf16*)(ws + 4 * SZ);   // [b,s,hc]  gated weighted average
  bf16* WqT   = (bf16*)(ws + 5 * SZ);
  bf16* WkT   = WqT + 65536;
  bf16* WvT   = WkT + 65536;
  bf16* WgT   = WvT + 65536;
  bf16* WoT   = WgT + 65536;

  prep_weights<<<(5 * 65536) / 256, 256, 0, stream>>>(
      Wq, Wk, Wv, Wg, Wo, WqT, WkT, WvT, WgT, WoT);

  dim3 gproj(NROW / 64, 4);
  proj_kernel<<<gproj, 256, 0, stream>>>(q_data, m_data, WqT, WkT, WvT, WgT,
                                         gbias, qb, kb, vT, gateT);

  const size_t ldsbytes = (size_t)LDS_ELEMS * sizeof(bf16);  // ~168.5 KiB
  attn_kernel<<<Bn * Hn, 256, ldsbytes, stream>>>(qb, kb, vT, gateT, bias,
                                                  nbias, wag);

  outproj_kernel<<<NROW / 64, 256, 0, stream>>>(wag, WoT, ob, out);
}
